// MatrixHybridRecurrentFFN_69123203662004
// MI455X (gfx1250) — matrix operand order repaired and checked by reading, not compiled, not run on hardware
//
#include <hip/hip_runtime.h>

// ---------------------------------------------------------------------------
// MatrixHybridRecurrentFFN for MI455X (gfx1250, wave32, WMMA bf16 path)
//   - all GEMMs: v_wmma_f32_16x16x32_bf16 (bf16 in, fp32 accum)
//   - fragments loaded as contiguous ds_load_b128 pairs (N-major B staging)
//   - A-tile staging via GLOBAL_LOAD_ASYNC_TO_LDS_B128 when available
//   - scan: chunked linear-attention reformulation, WMMA per chunk
// ---------------------------------------------------------------------------

typedef __attribute__((ext_vector_type(16))) __bf16 v16bf;
typedef __attribute__((ext_vector_type(8)))  float  v8f;

#define DEVI __device__ __forceinline__

constexpr int kB = 4, kS = 2048, kH = 1024, kM = 64, kLS = 2048;
constexpr int kTOK = kB * kS;  // 8192

#if defined(__has_builtin)
#if __has_builtin(__builtin_amdgcn_global_load_async_to_lds_b128) && \
    __has_builtin(__builtin_amdgcn_s_wait_asynccnt)
#define USE_ASYNC_LDS 1
#endif
#endif
#ifndef USE_ASYNC_LDS
#define USE_ASYNC_LDS 0
#endif

#if USE_ASYNC_LDS
// exact parameter type per hipcc diagnostic: int __vector_size(16), AS1 / AS3
typedef int i32x4v __attribute__((vector_size(16)));
typedef __attribute__((address_space(1))) i32x4v* as1_i32x4;
typedef __attribute__((address_space(3))) i32x4v* as3_i32x4;
#endif

DEVI unsigned short f2bfu(float f) {
  union { float f; unsigned u; } c; c.f = f;
  const unsigned u = c.u;
  return (unsigned short)((u + 0x7FFFu + ((u >> 16) & 1u)) >> 16);  // RNE
}
DEVI float bf2f(unsigned short u) {
  union { unsigned u; float f; } c; c.u = (unsigned)u << 16;
  return c.f;
}
DEVI float siluf(float v) { return v / (1.f + __expf(-v)); }

DEVI v8f wmma_bf16(v16bf a, v16bf b, v8f c) {
  return __builtin_amdgcn_wmma_f32_16x16x32_bf16(false, a, false, b, (short)0, c,
                                                 false, false);
}

DEVI v16bf make_frag(uint4 lo, uint4 hi) {
  union { uint4 q[2]; v16bf v; } u;
  u.q[0] = lo; u.q[1] = hi;
  return u.v;
}

// ---- WMMA fragment loaders (contiguous 16B LDS reads) --------------------
// A operand 16x32, source row-major: elem(m,k) = p[m*ld + k].
// Lane layout (ISA 7.12.2): m = lane&15; halves h=0..7 -> k=ko..ko+7,
// h=8..15 -> k=16+ko..23+ko, ko = (lane>=16 ? 8 : 0). Two contiguous chunks.
DEVI v16bf fragA_rm(const unsigned short* p, int ld, int lane) {
  const int m = lane & 15;
  const int ko = (lane & 16) ? 8 : 0;
  const unsigned short* r = p + m * ld + ko;
  return make_frag(*(const uint4*)r, *(const uint4*)(r + 16));
}
// B operand 32x16, source N-major (transposed): elem(k,n) = p[n*ld + k].
// Lane layout: n = lane&15; halves h=0..7 -> k=kb..kb+7,
// h=8..15 -> k=16+kb..23+kb, kb = (lane>=16 ? 8 : 0). Two contiguous chunks.
DEVI v16bf fragB_nm(const unsigned short* p, int ld, int lane) {
  const int n = lane & 15;
  const int kb = (lane & 16) ? 8 : 0;
  const unsigned short* r = p + n * ld + kb;
  return make_frag(*(const uint4*)r, *(const uint4*)(r + 16));
}

// ---------------------------------------------------------------------------
// fp32 -> bf16 converters
// ---------------------------------------------------------------------------
__global__ void cvt_bf16_kernel(const float* __restrict__ in,
                                unsigned short* __restrict__ out, int n) {
  const int i = blockIdx.x * blockDim.x + threadIdx.x;
  if (i < n) out[i] = f2bfu(in[i]);
}
__global__ void silu_cvt_kernel(const float* __restrict__ in,
                                unsigned short* __restrict__ out, int n) {
  const int i = blockIdx.x * blockDim.x + threadIdx.x;
  if (i < n) out[i] = f2bfu(siluf(in[i]));
}

// ---------------------------------------------------------------------------
// decay = sigmoid(x @ W_decay): one wave32 per token
// ---------------------------------------------------------------------------
__global__ __launch_bounds__(256) void decay_kernel(
    const float* __restrict__ x, const float* __restrict__ wd,
    float* __restrict__ dec) {
  const int tok = blockIdx.x * 8 + (threadIdx.x >> 5);
  const int lane = threadIdx.x & 31;
  const float* xr = x + (size_t)tok * kH;
  float s = 0.f;
  for (int h = lane; h < kH; h += 32) s = fmaf(xr[h], wd[h], s);
#pragma unroll
  for (int off = 16; off; off >>= 1) s += __shfl_down(s, off, 32);
  if (lane == 0) dec[tok] = 1.f / (1.f + __expf(-s));
}

// ---------------------------------------------------------------------------
// Generic tiled WMMA GEMM: C(MxN,f32 accum) = A(MxK,bf16) @ B(KxN,bf16)
// 256 threads = 8 waves (WR x WC), block tile BM x BN, BK = 32.
// A staged row-major (async-LDS when available); B staged N-major in LDS so
// B fragments are contiguous ds_load_b128 pairs.
// Requires M%BM==0, N%BN==0, K%32==0 (true for all calls here).
// ---------------------------------------------------------------------------
constexpr int EPI_F32 = 0, EPI_BF16 = 1, EPI_SILU_MUL_BF16 = 2, EPI_ADD_F32 = 3;

template <int BM, int BN, int WR, int WC, int EPI>
__global__ __launch_bounds__(256) void gemm_bf16_kernel(
    const unsigned short* __restrict__ A, const unsigned short* __restrict__ Bw,
    void* __restrict__ Cout, const unsigned short* __restrict__ aux,
    int N, int K) {
  static_assert(WR * WC == 8, "8 waves per block");
  constexpr int BK = 32;
  constexpr int WTM = BM / WR, WTN = BN / WC;
  constexpr int TM = WTM / 16, TN = WTN / 16;
  __shared__ __align__(16) unsigned short As[BM * BK];   // row-major [m][k]
  __shared__ __align__(16) unsigned short Bt[BN * BK];   // N-major  [n][k]

  const int tid = threadIdx.x;
  const int lane = tid & 31;
  const int wave = tid >> 5;
  const int wr = wave / WC, wc = wave % WC;
  const int m0 = blockIdx.x * BM;
  const int n0 = blockIdx.y * BN;

  v8f acc[TM][TN];
#pragma unroll
  for (int i = 0; i < TM; ++i)
#pragma unroll
    for (int j = 0; j < TN; ++j)
      acc[i][j] = v8f{0.f, 0.f, 0.f, 0.f, 0.f, 0.f, 0.f, 0.f};

  // B staging decomposition: thread -> (k-pair, 8-wide n group)
  constexpr int NGRP = BN / 8;
  const int kp = tid / NGRP;  // k-pair index, need kp < 16
  const int ng = tid % NGRP;

  for (int k0 = 0; k0 < K; k0 += BK) {
    // ---- stage A tile (BM x 32), 16B per thread per issue ----
#if USE_ASYNC_LDS
    for (int i = tid * 8; i < BM * BK; i += 256 * 8) {
      const int r = i / BK, c = i % BK;
      __builtin_amdgcn_global_load_async_to_lds_b128(
          (as1_i32x4)(void*)(A + (size_t)(m0 + r) * K + k0 + c),
          (as3_i32x4)(void*)&As[i], 0, 0);
    }
#else
    for (int i = tid * 8; i < BM * BK; i += 256 * 8) {
      const int r = i / BK, c = i % BK;
      *(uint4*)&As[i] = *(const uint4*)(A + (size_t)(m0 + r) * K + k0 + c);
    }
#endif
    // ---- stage B tile transposed: Bt[n][k], coalesced uint4 global loads,
    //      two K-rows packed into u32 LDS stores ----
    if (kp < 16) {
      const int k = 2 * kp;
      const int n = ng * 8;
      const uint4 r0 = *(const uint4*)(Bw + (size_t)(k0 + k) * N + n0 + n);
      const uint4 r1 = *(const uint4*)(Bw + (size_t)(k0 + k + 1) * N + n0 + n);
      const unsigned short* e0 = (const unsigned short*)&r0;
      const unsigned short* e1 = (const unsigned short*)&r1;
#pragma unroll
      for (int e = 0; e < 8; ++e)
        *(unsigned*)&Bt[(n + e) * BK + k] =
            (unsigned)e0[e] | ((unsigned)e1[e] << 16);
    }
    if (k0 + BK < K) {  // global_prefetch_b8 of next K-slice
      __builtin_prefetch(A + (size_t)(m0 + (tid >> 1)) * K + k0 + BK, 0, 1);
      __builtin_prefetch(Bw + (size_t)(k0 + BK + (tid >> 3)) * N + n0, 0, 1);
    }
#if USE_ASYNC_LDS
    __builtin_amdgcn_s_wait_asynccnt(0);
#endif
    __syncthreads();

    v16bf af[TM], bfv[TN];
#pragma unroll
    for (int i = 0; i < TM; ++i)
      af[i] = fragA_rm(&As[(wr * WTM + i * 16) * BK], BK, lane);
#pragma unroll
    for (int j = 0; j < TN; ++j)
      bfv[j] = fragB_nm(&Bt[(wc * WTN + j * 16) * BK], BK, lane);
#pragma unroll
    for (int i = 0; i < TM; ++i)
#pragma unroll
      for (int j = 0; j < TN; ++j)
        acc[i][j] = wmma_bf16(af[i], bfv[j], acc[i][j]);
    __syncthreads();
  }

  // epilogue: C/D layout: m = tile + r + (lane>=16 ? 8:0), n = lane&15
#pragma unroll
  for (int i = 0; i < TM; ++i)
#pragma unroll
    for (int j = 0; j < TN; ++j)
#pragma unroll
      for (int r = 0; r < 8; ++r) {
        const int m = m0 + wr * WTM + i * 16 + r + ((lane & 16) ? 8 : 0);
        const int n = n0 + wc * WTN + j * 16 + (lane & 15);
        const size_t idx = (size_t)m * N + n;
        const float v = acc[i][j][r];
        if constexpr (EPI == EPI_F32) {
          ((float*)Cout)[idx] = v;
        } else if constexpr (EPI == EPI_BF16) {
          ((unsigned short*)Cout)[idx] = f2bfu(v);
        } else if constexpr (EPI == EPI_SILU_MUL_BF16) {
          ((unsigned short*)Cout)[idx] = f2bfu(bf2f(aux[idx]) * siluf(v));
        } else {  // EPI_ADD_F32
          ((float*)Cout)[idx] = ((float*)Cout)[idx] + v;
        }
      }
}

// ---------------------------------------------------------------------------
// Chunked scan (T=16): one wave per batch; 64x64 fp32 state in 16 C-fragments.
//   state_t = d_t*state + (1-d_t)*(k v^T)/2
//   Let D_t = prod d. Within a chunk:
//     reads = [ (Q.*D_t) @ S0  +  mask(Qs @ Ks^T) @ V ] .* qr
//     with Ks row jr scaled by (1-d_j)/(2 D_j), Qs row t scaled by D_t
//     S_end = D_end * ( S0 + Ks^T @ V )
// LDS layouts chosen so every fragment load is contiguous:
//   Qs   [t][f]   row-major  (A operand)
//   Ks   [jr][f]  row-major == N-major for its use as Ks^T B-operand
//   Kst  [f][jr]  transposed (A operand for state update)
//   Vst  [n][jr]  N-major    (B operand, both uses)
//   Sbt  [n][m]   N-major    (B operand for cross term)
//   Aatt [t][jr]  row-major  (A operand)
// ---------------------------------------------------------------------------
__global__ __launch_bounds__(32) void scan_kernel(
    const float* __restrict__ keys, const float* __restrict__ vals,
    const float* __restrict__ ql, const float* __restrict__ qr,
    const float* __restrict__ decay, const float* __restrict__ init_state,
    float* __restrict__ reads) {
  constexpr int T = 16, NC = kS / T;
  const int b = blockIdx.x;
  const int lane = threadIdx.x & 31;
  const size_t tok0 = (size_t)b * kS;

  __shared__ __align__(16) unsigned short Qs[T * 64];
  __shared__ __align__(16) unsigned short Ks[32 * 64];
  __shared__ __align__(16) unsigned short Kst[64 * 32];
  __shared__ __align__(16) unsigned short Vst[64 * 32];
  __shared__ __align__(16) unsigned short Sbt[64 * 64];
  __shared__ __align__(16) unsigned short Aatt[T * 32];
  __shared__ float qrs[T * 64];
  __shared__ float dsh[T], coefq[T], coefk[T];

  v8f st[4][4];
#pragma unroll
  for (int mt = 0; mt < 4; ++mt)
#pragma unroll
    for (int nt = 0; nt < 4; ++nt)
#pragma unroll
      for (int r = 0; r < 8; ++r) {
        const int m = mt * 16 + r + ((lane & 16) ? 8 : 0);
        const int n = nt * 16 + (lane & 15);
        st[mt][nt][r] = init_state[m * 64 + n];
      }

  for (int c = 0; c < NC; ++c) {
    const int t0 = c * T;
    // state -> Sbt[n][m] as bf16 (paired-half packed stores)
#pragma unroll
    for (int mt = 0; mt < 4; ++mt)
#pragma unroll
      for (int nt = 0; nt < 4; ++nt) {
        const int n = nt * 16 + (lane & 15);
        const int mb = mt * 16 + ((lane & 16) ? 8 : 0);
#pragma unroll
        for (int r = 0; r < 8; r += 2) {
          const unsigned lo = f2bfu(st[mt][nt][r]);
          const unsigned hi = f2bfu(st[mt][nt][r + 1]);
          *(unsigned*)&Sbt[n * 64 + mb + r] = lo | (hi << 16);
        }
      }
    if (lane < T) dsh[lane] = decay[tok0 + t0 + lane];
    __syncthreads();
    if (lane == 0) {
      float D = 1.f;
      for (int j = 0; j < T; ++j) {
        const float d = dsh[j];
        D *= d;
        coefq[j] = D;
        coefk[j] = (1.f - d) * 0.5f / D;
      }
    }
    __syncthreads();
    // stage Q (decay-scaled) + qr
    for (int i = lane; i < T * 64; i += 32) {
      const int t = i >> 6, m = i & 63;
      Qs[i] = f2bfu(ql[(tok0 + t0 + t) * 64 + m] * coefq[t]);
      qrs[i] = qr[(tok0 + t0 + t) * 64 + m];
    }
    // stage K (decay-scaled) + V: jr = 2*j + r, row layout [r0(64) | r1(64)]
    for (int i = lane; i < 32 * 64; i += 32) {
      const int jr = i >> 6, f = i & 63;
      const int j = jr >> 1, rr = jr & 1;
      const size_t row = (tok0 + t0 + j) * 128 + rr * 64 + f;
      const unsigned short kv = f2bfu(keys[row] * coefk[j]);
      const unsigned short vv = f2bfu(vals[row]);
      Ks[jr * 64 + f] = kv;    // for Ks^T B-operand
      Kst[f * 32 + jr] = kv;   // for state-update A-operand
      Vst[f * 32 + jr] = vv;   // N-major V B-operand
    }
    __syncthreads();

    // cross reads: Qs(16x64) @ S0(64x64)
    v8f rd[4];
#pragma unroll
    for (int nt = 0; nt < 4; ++nt) {
      v8f a = v8f{0.f, 0.f, 0.f, 0.f, 0.f, 0.f, 0.f, 0.f};
#pragma unroll
      for (int kt = 0; kt < 2; ++kt)
        a = wmma_bf16(fragA_rm(&Qs[kt * 32], 64, lane),
                      fragB_nm(&Sbt[(nt * 16) * 64 + kt * 32], 64, lane), a);
      rd[nt] = a;
    }
    // Aatt = Qs @ Ks^T with causal mask (j <= t)
#pragma unroll
    for (int jt = 0; jt < 2; ++jt) {
      v8f a = v8f{0.f, 0.f, 0.f, 0.f, 0.f, 0.f, 0.f, 0.f};
#pragma unroll
      for (int kt = 0; kt < 2; ++kt)
        a = wmma_bf16(fragA_rm(&Qs[kt * 32], 64, lane),
                      fragB_nm(&Ks[(jt * 16) * 64 + kt * 32], 64, lane), a);
#pragma unroll
      for (int r = 0; r < 8; ++r) {
        const int t = r + ((lane & 16) ? 8 : 0);
        const int jr = jt * 16 + (lane & 15);
        const float v = ((jr >> 1) <= t) ? a[r] : 0.f;
        Aatt[t * 32 + jr] = f2bfu(v);
      }
    }
    __syncthreads();
    // inner reads: Aatt(16x32) @ V(32x64)
    {
      const v16bf af = fragA_rm(&Aatt[0], 32, lane);
#pragma unroll
      for (int nt = 0; nt < 4; ++nt)
        rd[nt] = wmma_bf16(af, fragB_nm(&Vst[(nt * 16) * 32], 32, lane), rd[nt]);
    }
    // reads epilogue: * qr, store
#pragma unroll
    for (int nt = 0; nt < 4; ++nt)
#pragma unroll
      for (int r = 0; r < 8; ++r) {
        const int t = r + ((lane & 16) ? 8 : 0);
        const int n = nt * 16 + (lane & 15);
        reads[(tok0 + t0 + t) * 64 + n] = rd[nt][r] * qrs[t * 64 + n];
      }
    // state update: S = Dend * (S0 + Ks^T @ V)
#pragma unroll
    for (int mt = 0; mt < 4; ++mt) {
      const v16bf am = fragA_rm(&Kst[(mt * 16) * 32], 32, lane);
#pragma unroll
      for (int nt = 0; nt < 4; ++nt)
        st[mt][nt] =
            wmma_bf16(am, fragB_nm(&Vst[(nt * 16) * 32], 32, lane), st[mt][nt]);
    }
    const float De = coefq[T - 1];
#pragma unroll
    for (int mt = 0; mt < 4; ++mt)
#pragma unroll
      for (int nt = 0; nt < 4; ++nt)
#pragma unroll
        for (int r = 0; r < 8; ++r) st[mt][nt][r] = st[mt][nt][r] * De;
    __syncthreads();
  }
}

// ---------------------------------------------------------------------------
// Host driver
// ---------------------------------------------------------------------------
extern "C" void kernel_launch(void* const* d_in, const int* in_sizes, int n_in,
                              void* d_out, int out_size, void* d_ws,
                              size_t ws_size, hipStream_t stream) {
  (void)in_sizes; (void)n_in; (void)out_size; (void)ws_size;
  const float* x     = (const float*)d_in[0];
  const float* Wdec  = (const float*)d_in[1];
  const float* Wkey  = (const float*)d_in[2];
  const float* Wval  = (const float*)d_in[3];
  const float* Wql   = (const float*)d_in[4];
  const float* Wqr   = (const float*)d_in[5];
  const float* Wrec  = (const float*)d_in[6];
  const float* Wup   = (const float*)d_in[7];
  const float* Wgate = (const float*)d_in[8];
  const float* Wdown = (const float*)d_in[9];
  const float* init  = (const float*)d_in[10];
  float* out = (float*)d_out;

  char* ws = (char*)d_ws;
  size_t off = 0;
  auto take = [&](size_t bytes) -> char* {
    char* p = ws + off;
    off = (off + bytes + 255) & ~(size_t)255;
    return p;
  };
  unsigned short* xb     = (unsigned short*)take((size_t)kTOK * kH * 2);
  unsigned short* wkb    = (unsigned short*)take((size_t)kH * 128 * 2);
  unsigned short* wvb    = (unsigned short*)take((size_t)kH * 128 * 2);
  unsigned short* wqlb   = (unsigned short*)take((size_t)kH * 64 * 2);
  unsigned short* wqrb   = (unsigned short*)take((size_t)kH * 64 * 2);
  unsigned short* wupb   = (unsigned short*)take((size_t)kH * kLS * 2);
  unsigned short* wgateb = (unsigned short*)take((size_t)kH * kLS * 2);
  unsigned short* wdownb = (unsigned short*)take((size_t)kLS * kH * 2);
  unsigned short* wrecb  = (unsigned short*)take((size_t)kM * kH * 2);
  float* keysf = (float*)take((size_t)kTOK * 128 * 4);
  float* valsf = (float*)take((size_t)kTOK * 128 * 4);
  float* qlf   = (float*)take((size_t)kTOK * 64 * 4);
  float* qrf   = (float*)take((size_t)kTOK * 64 * 4);
  float* decf  = (float*)take((size_t)kTOK * 4);
  float* readsf = (float*)take((size_t)kTOK * 64 * 4);
  unsigned short* rsb  = (unsigned short*)take((size_t)kTOK * 64 * 2);
  unsigned short* upb  = (unsigned short*)take((size_t)kTOK * kLS * 2);
  unsigned short* hidb = (unsigned short*)take((size_t)kTOK * kLS * 2);

  auto cvt = [&](const float* src, unsigned short* dst, int n) {
    cvt_bf16_kernel<<<(n + 255) / 256, 256, 0, stream>>>(src, dst, n);
  };
  cvt(x, xb, kTOK * kH);
  cvt(Wkey, wkb, kH * 128);
  cvt(Wval, wvb, kH * 128);
  cvt(Wql, wqlb, kH * 64);
  cvt(Wqr, wqrb, kH * 64);
  cvt(Wup, wupb, kH * kLS);
  cvt(Wgate, wgateb, kH * kLS);
  cvt(Wdown, wdownb, kLS * kH);
  cvt(Wrec, wrecb, kM * kH);

  decay_kernel<<<kTOK / 8, 256, 0, stream>>>(x, Wdec, decf);

  // projections: x(8192x1024) @ W -> fp32
  gemm_bf16_kernel<128, 64, 4, 2, EPI_F32><<<dim3(kTOK / 128, 2), 256, 0, stream>>>(
      xb, wkb, keysf, nullptr, 128, kH);
  gemm_bf16_kernel<128, 64, 4, 2, EPI_F32><<<dim3(kTOK / 128, 2), 256, 0, stream>>>(
      xb, wvb, valsf, nullptr, 128, kH);
  gemm_bf16_kernel<128, 64, 4, 2, EPI_F32><<<dim3(kTOK / 128, 1), 256, 0, stream>>>(
      xb, wqlb, qlf, nullptr, 64, kH);
  gemm_bf16_kernel<128, 64, 4, 2, EPI_F32><<<dim3(kTOK / 128, 1), 256, 0, stream>>>(
      xb, wqrb, qrf, nullptr, 64, kH);

  // chunked recurrence (one wave32 per batch)
  scan_kernel<<<kB, 32, 0, stream>>>(keysf, valsf, qlf, qrf, decf, init, readsf);
  silu_cvt_kernel<<<(kTOK * 64 + 255) / 256, 256, 0, stream>>>(readsf, rsb,
                                                               kTOK * 64);

  // FFN: up, gate (fused silu*up epilogue), down -> out
  gemm_bf16_kernel<128, 128, 2, 4, EPI_BF16>
      <<<dim3(kTOK / 128, kLS / 128), 256, 0, stream>>>(xb, wupb, upb, nullptr,
                                                        kLS, kH);
  gemm_bf16_kernel<128, 128, 2, 4, EPI_SILU_MUL_BF16>
      <<<dim3(kTOK / 128, kLS / 128), 256, 0, stream>>>(xb, wgateb, hidb, upb,
                                                        kLS, kH);
  gemm_bf16_kernel<128, 128, 2, 4, EPI_F32>
      <<<dim3(kTOK / 128, kH / 128), 256, 0, stream>>>(hidb, wdownb, out,
                                                       nullptr, kH, kLS);
  // recurrent branch: out += silu(reads) @ W_rec_out
  gemm_bf16_kernel<128, 128, 2, 4, EPI_ADD_F32>
      <<<dim3(kTOK / 128, kH / 128), 256, 0, stream>>>(rsb, wrecb, out, nullptr,
                                                       kH, kM);
}
